// ViTTM_HMR_71975061946682
// MI455X (gfx1250) — compile-verified, matrix operand order checked by reading, not run on hardware
//
#include <hip/hip_runtime.h>
#include <hip/hip_bf16.h>
#include <math.h>

// ---------------- model dims ----------------
#define B_SZ    32
#define D_MODEL 384
#define NHEADS  6
#define HD      64
#define LPTOK   256     // process tokens
#define MEMTOK  256     // memory tokens
#define MLPDIM  1536
#define DEPTH_N 6
#define TGT_TOK 1024    // target tokens per image (512/16)^2

typedef __attribute__((ext_vector_type(16))) __bf16 v16bf;
typedef __attribute__((ext_vector_type(8)))  __bf16 v8bf;
typedef __attribute__((ext_vector_type(8)))  float  v8f;

__device__ __forceinline__ unsigned short f32_to_bf16(float f) {
  unsigned int u = __builtin_bit_cast(unsigned int, f);
  u += 0x7fffu + ((u >> 16) & 1u);          // round-to-nearest-even
  return (unsigned short)(u >> 16);
}
__device__ __forceinline__ float gelu_f(float x) {
  return 0.5f * x * (1.0f + erff(x * 0.70710678118654752f));
}
__device__ __forceinline__ float elu1_f(float x) {      // elu(x)+1
  return x > 0.0f ? x + 1.0f : expf(x);
}

// ---------------- WMMA GEMM ----------------
// C[M,N] = epilogue( A * B + bias ) + res     (bf16 inputs, f32 accum)
// aT : A stored [K,M] (element (m,k) at A[k*lda+m])
// bNK: B stored [N,K] (element (k,n) at B[n*ldb+k]); else [K,N]
// act: 0 none, 1 gelu, 2 elu+1
// batched over blockIdx.z with z = bb*nh + hh and per-(bb,hh) pointer strides
struct GemmArgs {
  const float* A; const float* B; float* C;
  const float* bias; const float* res;
  int M, N, K, lda, ldb, ldc, aT, bNK, act, nh;
  long long sAb, sAh, sBb, sBh, sCb, sCh, sRb, sRh;
};

#define TK 32

__device__ __forceinline__ v16bf load_frag(const unsigned short* row, int khi) {
  v8bf lo = *(const v8bf*)(row + khi);        // ds_load_b128
  v8bf hi = *(const v8bf*)(row + khi + 16);   // ds_load_b128
  v16bf r;
#pragma unroll
  for (int i = 0; i < 8; ++i) { r[i] = lo[i]; r[i + 8] = hi[i]; }
  return r;
}

__device__ __forceinline__ unsigned long long pack4bf(float4 v) {
  return (unsigned long long)f32_to_bf16(v.x)
       | ((unsigned long long)f32_to_bf16(v.y) << 16)
       | ((unsigned long long)f32_to_bf16(v.z) << 32)
       | ((unsigned long long)f32_to_bf16(v.w) << 48);
}

// TMN: square block tile (64 or 128); 4 wave32 waves; wave tile = TMN/2 x TMN/2
template <int TMN>
__global__ __launch_bounds__(128) void gemm_wmma_bf16(GemmArgs g) {
  constexpr int MT = TMN / 32;    // 16x16 tiles per wave in m
  constexpr int NT = TMN / 32;    // and in n
  const int z  = blockIdx.z;
  const int bb = z / g.nh, hh = z % g.nh;
  const float* A = g.A + bb * g.sAb + hh * g.sAh;
  const float* B = g.B + bb * g.sBb + hh * g.sBh;
  float*       C = g.C + bb * g.sCb + hh * g.sCh;
  const float* R = g.res ? (g.res + bb * g.sRb + hh * g.sRh) : nullptr;

  __shared__ unsigned short As[TMN][TK];   // [m][k] bf16 bits
  __shared__ unsigned short Bs[TMN][TK];   // [n][k] bf16 bits (B^T)

  const int m0   = blockIdx.x * TMN;
  const int n0   = blockIdx.y * TMN;
  const int tid  = threadIdx.x;     // 128 threads = 4 wave32 waves
  const int wave = tid >> 5;
  const int lane = tid & 31;
  const int wm   = (wave & 1) * (TMN / 2);
  const int wn   = (wave >> 1) * (TMN / 2);
  const int lrow = lane & 15;
  const int khi  = (lane >> 4) * 8;   // ISA 16-bit A/B per-lane K layout

  v8f acc[MT][NT];
#pragma unroll
  for (int mi = 0; mi < MT; ++mi)
#pragma unroll
    for (int ni = 0; ni < NT; ++ni)
#pragma unroll
      for (int r = 0; r < 8; ++r) acc[mi][ni][r] = 0.0f;

  for (int k0 = 0; k0 < g.K; k0 += TK) {
    // ---- stage A tile (TMN x 32) as bf16 ----
    if (!g.aT) {
      // rows contiguous in k: float4 global load -> packed ds_store_b64
#pragma unroll
      for (int t = tid; t < TMN * 8; t += 128) {
        int m = t >> 3, kq = (t & 7) * 4;
        float4 v = *(const float4*)(A + (long long)(m0 + m) * g.lda + (k0 + kq));
        *(unsigned long long*)&As[m][kq] = pack4bf(v);
      }
    } else {
      // A stored [K,M]: contiguous in m -> float4 load, 4 scattered b16 stores
#pragma unroll
      for (int t = tid; t < 32 * (TMN / 4); t += 128) {
        int kk = t & 31, mg = (t >> 5) * 4;
        float4 v = *(const float4*)(A + (long long)(k0 + kk) * g.lda + (m0 + mg));
        As[mg + 0][kk] = f32_to_bf16(v.x);
        As[mg + 1][kk] = f32_to_bf16(v.y);
        As[mg + 2][kk] = f32_to_bf16(v.z);
        As[mg + 3][kk] = f32_to_bf16(v.w);
      }
    }
    // ---- stage B tile transposed -> Bs[n][k] ----
    if (g.bNK) {
#pragma unroll
      for (int t = tid; t < TMN * 8; t += 128) {
        int n = t >> 3, kq = (t & 7) * 4;
        float4 v = *(const float4*)(B + (long long)(n0 + n) * g.ldb + (k0 + kq));
        *(unsigned long long*)&Bs[n][kq] = pack4bf(v);
      }
    } else {
#pragma unroll
      for (int t = tid; t < 32 * (TMN / 4); t += 128) {
        int kk = t & 31, ng = (t >> 5) * 4;
        float4 v = *(const float4*)(B + (long long)(k0 + kk) * g.ldb + (n0 + ng));
        Bs[ng + 0][kk] = f32_to_bf16(v.x);
        Bs[ng + 1][kk] = f32_to_bf16(v.y);
        Bs[ng + 2][kk] = f32_to_bf16(v.z);
        Bs[ng + 3][kk] = f32_to_bf16(v.w);
      }
    }
    if (k0 + TK < g.K) {   // global_prefetch_b8 next K tile
      int m = tid >> 5, kk = tid & 31;
      __builtin_prefetch(g.aT ? &A[(long long)(k0 + TK + kk) * g.lda + (m0 + m)]
                              : &A[(long long)(m0 + m) * g.lda + (k0 + TK + kk)], 0, 1);
    }
    __syncthreads();

    // ---- compute: hoist B fragments, MT+NT frag loads for MT*NT WMMAs ----
    v16bf bfr[NT];
#pragma unroll
    for (int ni = 0; ni < NT; ++ni)
      bfr[ni] = load_frag(&Bs[wn + ni * 16 + lrow][0], khi);
#pragma unroll
    for (int mi = 0; mi < MT; ++mi) {
      v16bf af = load_frag(&As[wm + mi * 16 + lrow][0], khi);
#pragma unroll
      for (int ni = 0; ni < NT; ++ni)
        acc[mi][ni] = __builtin_amdgcn_wmma_f32_16x16x32_bf16(
            false, af, false, bfr[ni], (short)0, acc[mi][ni], false, false);
    }
    __syncthreads();
  }

  // D layout: lanes 0-15 -> n=lane, m=r ; lanes 16-31 -> n=lane-16, m=r+8
  const int cn  = lane & 15;
  const int cm0 = (lane >> 4) * 8;
#pragma unroll
  for (int mi = 0; mi < MT; ++mi)
#pragma unroll
    for (int ni = 0; ni < NT; ++ni)
#pragma unroll
      for (int r = 0; r < 8; ++r) {
        int m = m0 + wm + mi * 16 + cm0 + r;
        int n = n0 + wn + ni * 16 + cn;
        float v = acc[mi][ni][r];
        if (g.bias) v += g.bias[n];
        if (g.act == 1)      v = gelu_f(v);
        else if (g.act == 2) v = elu1_f(v);
        if (R) v += R[(long long)m * g.ldc + n];
        C[(long long)m * g.ldc + n] = v;
      }
}

// ---------------- elementwise / reduction kernels ----------------

__global__ __launch_bounds__(128) void ln_kernel(const float* x, float* y,
                                                 const float* g, const float* b, int D) {
  const long long row = blockIdx.x;
  const float* xr = x + row * D;
  float* yr = y + row * D;
  const int tid = threadIdx.x;
  __shared__ float s1[128], s2[128];
  float s = 0.f, ss = 0.f;
  for (int i = tid; i < D; i += 128) { float v = xr[i]; s += v; ss += v * v; }
  s1[tid] = s; s2[tid] = ss; __syncthreads();
  for (int st = 64; st > 0; st >>= 1) {
    if (tid < st) { s1[tid] += s1[tid + st]; s2[tid] += s2[tid + st]; }
    __syncthreads();
  }
  float mean = s1[0] / D;
  float var  = s2[0] / D - mean * mean;
  float inv  = rsqrtf(var + 1e-5f);
  for (int i = tid; i < D; i += 128)
    yr[i] = (xr[i] - mean) * inv * g[i] + b[i];
}

__global__ __launch_bounds__(256) void softmax256(float* S, float scale) {
  const long long row = blockIdx.x;
  float* r = S + row * 256;
  const int tid = threadIdx.x;
  __shared__ float sh[256];
  float v = r[tid] * scale;
  sh[tid] = v; __syncthreads();
  for (int st = 128; st > 0; st >>= 1) {
    if (tid < st) sh[tid] = fmaxf(sh[tid], sh[tid + st]);
    __syncthreads();
  }
  float mx = sh[0]; __syncthreads();
  float e = expf(v - mx);
  sh[tid] = e; __syncthreads();
  for (int st = 128; st > 0; st >>= 1) {
    if (tid < st) sh[tid] += sh[tid + st];
    __syncthreads();
  }
  r[tid] = e / sh[0];
}

__global__ void ksum_kernel(const float* k, float* ks, int Ntok) {
  int idx = blockIdx.x * blockDim.x + threadIdx.x;     // B*H*64
  if (idx >= B_SZ * NHEADS * HD) return;
  int d = idx & 63, h = (idx >> 6) % NHEADS, b = idx / (HD * NHEADS);
  const float* kp = k + (long long)b * Ntok * D_MODEL + h * HD + d;
  float s = 0.f;
  for (int n = 0; n < Ntok; ++n) s += kp[(long long)n * D_MODEL];
  ks[idx] = s;
}

__global__ void zscale_kernel(const float* q, const float* ks, float* o, int Nq) {
  int idx = blockIdx.x * blockDim.x + threadIdx.x;     // B*H*Nq
  if (idx >= B_SZ * NHEADS * Nq) return;
  int n = idx % Nq, h = (idx / Nq) % NHEADS, b = idx / (Nq * NHEADS);
  const float* qp = q + ((long long)b * Nq + n) * D_MODEL + h * HD;
  const float* kp = ks + (b * NHEADS + h) * HD;
  float dot = 0.f;
  for (int d = 0; d < HD; ++d) dot += qp[d] * kp[d];
  float zz = 1.0f / (dot + 1e-6f);
  float* op = o + ((long long)b * Nq + n) * D_MODEL + h * HD;
  for (int d = 0; d < HD; ++d) op[d] *= zz;
}

__global__ __launch_bounds__(256) void mask_kernel(const float* noise, float* proc,
                                                   const float* mask_token) {
  int b = blockIdx.x, j = threadIdx.x;
  __shared__ float ns[LPTOK];
  ns[j] = noise[b * LPTOK + j];
  __syncthreads();
  float nj = ns[j];
  int rank = 0;
  for (int i = 0; i < LPTOK; ++i) {
    float ni = ns[i];
    rank += (ni < nj) || (ni == nj && i < j);
  }
  if (rank >= 64) {    // LEN_KEEP
    float* p = proc + ((long long)b * LPTOK + j) * D_MODEL;
    for (int d = 0; d < D_MODEL; ++d) p[d] = mask_token[d];
  }
}

__global__ void addpos_kernel(float* proc, const float* pos) {
  long long i = (long long)blockIdx.x * blockDim.x + threadIdx.x;
  if (i >= (long long)B_SZ * LPTOK * D_MODEL) return;
  proc[i] += pos[i % (LPTOK * D_MODEL)];
}

__global__ void bcast_mem_kernel(float* mem, const float* bank) {
  long long i = (long long)blockIdx.x * blockDim.x + threadIdx.x;
  if (i >= (long long)B_SZ * MEMTOK * D_MODEL) return;
  mem[i] = bank[i % (MEMTOK * D_MODEL)];
}

__global__ void im2col_tgt(const float* img, float* out) {   // [32768, 768]
  long long idx = (long long)blockIdx.x * blockDim.x + threadIdx.x;
  if (idx >= (long long)B_SZ * TGT_TOK * 768) return;
  int kk = (int)(idx % 768);
  long long row = idx / 768;
  int b = (int)(row / TGT_TOK), p = (int)(row % TGT_TOK);
  int ph = p / 32, pw = p % 32;
  int c = kk / 256, rr = kk % 256, i2 = rr / 16, j2 = rr % 16;
  out[idx] = img[(((long long)b * 3 + c) * 512 + ph * 16 + i2) * 512 + pw * 16 + j2];
}

__global__ void im2col_proc(const float* img, float* out) {  // [8192, 3072]
  long long idx = (long long)blockIdx.x * blockDim.x + threadIdx.x;
  if (idx >= (long long)B_SZ * LPTOK * 3072) return;
  int kk = (int)(idx % 3072);
  long long row = idx / 3072;
  int b = (int)(row / LPTOK), p = (int)(row % LPTOK);
  int ph = p / 16, pw = p % 16;
  int c = kk / 1024, rr = kk % 1024, i2 = rr / 32, j2 = rr % 32;
  out[idx] = img[(((long long)b * 3 + c) * 512 + ph * 32 + i2) * 512 + pw * 32 + j2];
}

__global__ void repack_ct(const float* w, float* out, int ki, int kj) {
  int idx = blockIdx.x * blockDim.x + threadIdx.x;   // 384*384
  if (idx >= D_MODEL * D_MODEL) return;
  out[idx] = w[(long long)idx * 4 + ki * 2 + kj];    // (c,d,ki,kj)
}

__global__ void scatter_ct(const float* tmp, float* yt, const float* bias, int ki, int kj) {
  long long idx = (long long)blockIdx.x * blockDim.x + threadIdx.x;  // B*256*384
  if (idx >= (long long)B_SZ * LPTOK * D_MODEL) return;
  int d = (int)(idx % D_MODEL);
  long long row = idx / D_MODEL;
  int b = (int)(row / LPTOK), p = (int)(row % LPTOK);
  int i2 = p / 16, j2 = p % 16;
  int tok = (2 * i2 + ki) * 32 + (2 * j2 + kj);
  yt[((long long)b * TGT_TOK + tok) * D_MODEL + d] = tmp[idx] + bias[d];
}

__global__ __launch_bounds__(256) void gn_gelu(float* yt, const float* gg, const float* gb) {
  int bg = blockIdx.x;                 // B*32 groups
  int b = bg / 32, g2 = bg % 32;
  const int CNT = 12 * TGT_TOK;        // 12 channels * 1024 pixels
  const int tid = threadIdx.x;
  __shared__ float s1[256], s2[256];
  float s = 0.f, ss = 0.f;
  for (int t = tid; t < CNT; t += 256) {
    int tok = t / 12, c = g2 * 12 + t % 12;
    float v = yt[((long long)b * TGT_TOK + tok) * D_MODEL + c];
    s += v; ss += v * v;
  }
  s1[tid] = s; s2[tid] = ss; __syncthreads();
  for (int st = 128; st > 0; st >>= 1) {
    if (tid < st) { s1[tid] += s1[tid + st]; s2[tid] += s2[tid + st]; }
    __syncthreads();
  }
  float mean = s1[0] / CNT;
  float var  = s2[0] / CNT - mean * mean;
  float inv  = rsqrtf(var + 1e-5f);
  __syncthreads();
  for (int t = tid; t < CNT; t += 256) {
    int tok = t / 12, c = g2 * 12 + t % 12;
    long long a = ((long long)b * TGT_TOK + tok) * D_MODEL + c;
    float v = (yt[a] - mean) * inv * gg[c] + gb[c];
    yt[a] = gelu_f(v);
  }
}

__global__ void zero1(float* p) { if (blockIdx.x == 0 && threadIdx.x == 0) p[0] = 0.0f; }

__global__ __launch_bounds__(128) void cosloss(const float* pred, const float* tgt, float* out) {
  const long long row = blockIdx.x;      // B*1024 rows
  const int tid = threadIdx.x;
  __shared__ float sd[128], sa[128], sb[128];
  float dot = 0.f, na = 0.f, nb = 0.f;
  for (int i = tid; i < D_MODEL; i += 128) {
    float a = pred[row * D_MODEL + i], b = tgt[row * D_MODEL + i];
    dot += a * b; na += a * a; nb += b * b;
  }
  sd[tid] = dot; sa[tid] = na; sb[tid] = nb; __syncthreads();
  for (int st = 64; st > 0; st >>= 1) {
    if (tid < st) { sd[tid] += sd[tid + st]; sa[tid] += sa[tid + st]; sb[tid] += sb[tid + st]; }
    __syncthreads();
  }
  if (tid == 0) {
    float denom = fmaxf(sqrtf(sa[0]), 1e-8f) * fmaxf(sqrtf(sb[0]), 1e-8f);
    atomicAdd(out, -(sd[0] / denom) / (float)(B_SZ * TGT_TOK));
  }
}

// ---------------- host orchestration ----------------
static void launch_gemm(hipStream_t s, const float* A, const float* B, float* C,
                        const float* bias, const float* res,
                        int M, int N, int K, int lda, int ldb, int ldc,
                        int aT, int bNK, int act,
                        int batches = 1, int nh = 1,
                        long long sAb = 0, long long sAh = 0,
                        long long sBb = 0, long long sBh = 0,
                        long long sCb = 0, long long sCh = 0,
                        long long sRb = 0, long long sRh = 0) {
  GemmArgs g{A, B, C, bias, res, M, N, K, lda, ldb, ldc, aT, bNK, act, nh,
             sAb, sAh, sBb, sBh, sCb, sCh, sRb, sRh};
  if ((M % 128 == 0) && (N % 128 == 0)) {
    dim3 grid(M / 128, N / 128, batches);
    gemm_wmma_bf16<128><<<grid, 128, 0, s>>>(g);
  } else {
    dim3 grid(M / 64, N / 64, batches);
    gemm_wmma_bf16<64><<<grid, 128, 0, s>>>(g);
  }
}

extern "C" void kernel_launch(void* const* d_in, const int* in_sizes, int n_in,
                              void* d_out, int out_size, void* d_ws, size_t ws_size,
                              hipStream_t stream) {
  (void)in_sizes; (void)n_in; (void)out_size; (void)ws_size;
  const float* process_view = (const float*)d_in[0];
  const float* memory_view  = (const float*)d_in[1];
  const float* noise        = (const float*)d_in[2];
  const float* te_w = (const float*)d_in[3];  const float* te_b = (const float*)d_in[4];
  const float* pe_w = (const float*)d_in[5];  const float* pe_b = (const float*)d_in[6];
  const float* memory_bank = (const float*)d_in[7];
  const float* pos_embed   = (const float*)d_in[8];
  const float* mask_token  = (const float*)d_in[9];
  const float* ln_r_g = (const float*)d_in[10]; const float* ln_r_b = (const float*)d_in[11];
  const float* r_q_w = (const float*)d_in[12];
  const float* r_k_w = (const float*)d_in[13];
  const float* r_v_w = (const float*)d_in[14];
  const float* r_o_w = (const float*)d_in[15]; const float* r_o_b = (const float*)d_in[16];
  const float* ln_p_g = (const float*)d_in[17]; const float* ln_p_b = (const float*)d_in[18];
  const float* a_q_w = (const float*)d_in[19]; const float* a_q_b = (const float*)d_in[20];
  const float* a_k_w = (const float*)d_in[21]; const float* a_k_b = (const float*)d_in[22];
  const float* a_v_w = (const float*)d_in[23]; const float* a_v_b = (const float*)d_in[24];
  const float* a_o_w = (const float*)d_in[25]; const float* a_o_b = (const float*)d_in[26];
  const float* ln_w_g = (const float*)d_in[27]; const float* ln_w_b = (const float*)d_in[28];
  const float* w_q_w = (const float*)d_in[29];
  const float* w_k_w = (const float*)d_in[30];
  const float* w_v_w = (const float*)d_in[31];
  const float* w_o_w = (const float*)d_in[32]; const float* w_o_b = (const float*)d_in[33];
  const float* ln_mp_g = (const float*)d_in[34]; const float* ln_mp_b = (const float*)d_in[35];
  const float* mp_w1 = (const float*)d_in[36]; const float* mp_b1 = (const float*)d_in[37];
  const float* mp_w2 = (const float*)d_in[38]; const float* mp_b2 = (const float*)d_in[39];
  const float* ln_mm_g = (const float*)d_in[40]; const float* ln_mm_b = (const float*)d_in[41];
  const float* mm_w1 = (const float*)d_in[42]; const float* mm_b1 = (const float*)d_in[43];
  const float* mm_w2 = (const float*)d_in[44]; const float* mm_b2 = (const float*)d_in[45];
  const float* norm_g = (const float*)d_in[46]; const float* norm_b = (const float*)d_in[47];
  const float* dec_ct_w = (const float*)d_in[48]; const float* dec_ct_b = (const float*)d_in[49];
  const float* gn_g = (const float*)d_in[50]; const float* gn_b = (const float*)d_in[51];
  const float* dec_c_w = (const float*)d_in[52]; const float* dec_c_b = (const float*)d_in[53];

  float* out  = (float*)d_out;
  float* loss = out;
  float* pred = out + 1;
  float* target = out + 1 + (long long)B_SZ * TGT_TOK * D_MODEL;

  // workspace carve-up (f32)
  float* ws = (float*)d_ws;
  size_t off = 0;
  auto alloc = [&](size_t n) { float* p = ws + off; off += (n + 63) & ~(size_t)63; return p; };
  const size_t ROWS_P = (size_t)B_SZ * LPTOK;           // 8192
  float* big   = alloc(25165824);                       // im2col / scores / mlp-hidden
  float* proc  = alloc(ROWS_P * D_MODEL);
  float* memb  = alloc(ROWS_P * D_MODEL);
  float* lnb   = alloc(ROWS_P * D_MODEL);
  float* qb    = alloc(ROWS_P * D_MODEL);
  float* kb    = alloc(ROWS_P * D_MODEL);
  float* vb    = alloc(ROWS_P * D_MODEL);
  float* ob    = alloc(ROWS_P * D_MODEL);
  float* kvb   = alloc((size_t)B_SZ * NHEADS * HD * HD);
  float* ksumb = alloc((size_t)B_SZ * NHEADS * HD);
  float* wpack = alloc((size_t)D_MODEL * D_MODEL);
  float* cttmp = alloc(ROWS_P * D_MODEL);
  float* yt    = alloc((size_t)B_SZ * TGT_TOK * D_MODEL);

  const long long sTok = (long long)LPTOK * D_MODEL;    // per-batch act stride
  const long long sKV  = (long long)NHEADS * HD * HD;
  const long long sSc  = (long long)NHEADS * LPTOK * LPTOK;

  auto blocks = [](long long n, int t) { return (unsigned)((n + t - 1) / t); };

  // ---- patchify target (no-grad branch): im2col + GEMM ----
  im2col_tgt<<<blocks((long long)B_SZ * TGT_TOK * 768, 256), 256, 0, stream>>>(memory_view, big);
  launch_gemm(stream, big, te_w, target, te_b, nullptr,
              B_SZ * TGT_TOK, D_MODEL, 768, 768, 768, D_MODEL, 0, /*bNK=*/1, 0);

  // ---- patchify process + pos embed + masking ----
  im2col_proc<<<blocks((long long)B_SZ * LPTOK * 3072, 256), 256, 0, stream>>>(process_view, big);
  launch_gemm(stream, big, pe_w, proc, pe_b, nullptr,
              (int)ROWS_P, D_MODEL, 3072, 3072, 3072, D_MODEL, 0, /*bNK=*/1, 0);
  addpos_kernel<<<blocks((long long)ROWS_P * D_MODEL, 256), 256, 0, stream>>>(proc, pos_embed);
  mask_kernel<<<B_SZ, LPTOK, 0, stream>>>(noise, proc, mask_token);
  bcast_mem_kernel<<<blocks((long long)ROWS_P * D_MODEL, 256), 256, 0, stream>>>(memb, memory_bank);

  // ---- transformer layers ----
  for (int i = 0; i < DEPTH_N; ++i) {
    const long long wO  = (long long)i * D_MODEL * D_MODEL;
    const long long bO  = (long long)i * D_MODEL;
    const long long w1O = (long long)i * D_MODEL * MLPDIM;
    const long long b1O = (long long)i * MLPDIM;
    const long long w2O = (long long)i * MLPDIM * D_MODEL;

    // ===== READ: proc += lin_xattn(ln(proc), mem) =====
    ln_kernel<<<(unsigned)ROWS_P, 128, 0, stream>>>(proc, lnb, ln_r_g + bO, ln_r_b + bO, D_MODEL);
    launch_gemm(stream, lnb,  r_q_w + wO, qb, nullptr, nullptr,
                (int)ROWS_P, D_MODEL, D_MODEL, D_MODEL, D_MODEL, D_MODEL, 0, 0, /*elu1*/2);
    launch_gemm(stream, memb, r_k_w + wO, kb, nullptr, nullptr,
                (int)ROWS_P, D_MODEL, D_MODEL, D_MODEL, D_MODEL, D_MODEL, 0, 0, 2);
    launch_gemm(stream, memb, r_v_w + wO, vb, nullptr, nullptr,
                (int)ROWS_P, D_MODEL, D_MODEL, D_MODEL, D_MODEL, D_MODEL, 0, 0, 0);
    // kv[b,h] = k^T v  (64x64, K=256)
    launch_gemm(stream, kb, vb, kvb, nullptr, nullptr,
                HD, HD, MEMTOK, D_MODEL, D_MODEL, HD, /*aT=*/1, 0, 0,
                B_SZ * NHEADS, NHEADS, sTok, HD, sTok, HD, sKV, (long long)HD * HD);
    ksum_kernel<<<blocks(B_SZ * NHEADS * HD, 256), 256, 0, stream>>>(kb, ksumb, MEMTOK);
    // o = q @ kv  (256x64, K=64)
    launch_gemm(stream, qb, kvb, ob, nullptr, nullptr,
                LPTOK, HD, HD, D_MODEL, HD, D_MODEL, 0, 0, 0,
                B_SZ * NHEADS, NHEADS, sTok, HD, sKV, (long long)HD * HD, sTok, HD);
    zscale_kernel<<<blocks((long long)B_SZ * NHEADS * LPTOK, 256), 256, 0, stream>>>(qb, ksumb, ob, LPTOK);
    launch_gemm(stream, ob, r_o_w + wO, proc, r_o_b + bO, proc,
                (int)ROWS_P, D_MODEL, D_MODEL, D_MODEL, D_MODEL, D_MODEL, 0, 0, 0);

    // ===== self-attention MHA on proc =====
    ln_kernel<<<(unsigned)ROWS_P, 128, 0, stream>>>(proc, lnb, ln_p_g + bO, ln_p_b + bO, D_MODEL);
    launch_gemm(stream, lnb, a_q_w + wO, qb, a_q_b + bO, nullptr,
                (int)ROWS_P, D_MODEL, D_MODEL, D_MODEL, D_MODEL, D_MODEL, 0, 0, 0);
    launch_gemm(stream, lnb, a_k_w + wO, kb, a_k_b + bO, nullptr,
                (int)ROWS_P, D_MODEL, D_MODEL, D_MODEL, D_MODEL, D_MODEL, 0, 0, 0);
    launch_gemm(stream, lnb, a_v_w + wO, vb, a_v_b + bO, nullptr,
                (int)ROWS_P, D_MODEL, D_MODEL, D_MODEL, D_MODEL, D_MODEL, 0, 0, 0);
    // scores = q @ k^T (256x256, K=64), B in [N,K] layout
    launch_gemm(stream, qb, kb, big, nullptr, nullptr,
                LPTOK, LPTOK, HD, D_MODEL, D_MODEL, LPTOK, 0, /*bNK=*/1, 0,
                B_SZ * NHEADS, NHEADS, sTok, HD, sTok, HD, sSc, (long long)LPTOK * LPTOK);
    softmax256<<<(unsigned)(B_SZ * NHEADS * LPTOK), 256, 0, stream>>>(big, 0.125f);
    // o = attn @ v (256x64, K=256)
    launch_gemm(stream, big, vb, ob, nullptr, nullptr,
                LPTOK, HD, LPTOK, LPTOK, D_MODEL, D_MODEL, 0, 0, 0,
                B_SZ * NHEADS, NHEADS, sSc, (long long)LPTOK * LPTOK, sTok, HD, sTok, HD);
    launch_gemm(stream, ob, a_o_w + wO, proc, a_o_b + bO, proc,
                (int)ROWS_P, D_MODEL, D_MODEL, D_MODEL, D_MODEL, D_MODEL, 0, 0, 0);

    // ===== MLP (proc) =====
    ln_kernel<<<(unsigned)ROWS_P, 128, 0, stream>>>(proc, lnb, ln_mp_g + bO, ln_mp_b + bO, D_MODEL);
    launch_gemm(stream, lnb, mp_w1 + w1O, big, mp_b1 + b1O, nullptr,
                (int)ROWS_P, MLPDIM, D_MODEL, D_MODEL, MLPDIM, MLPDIM, 0, 0, /*gelu*/1);
    launch_gemm(stream, big, mp_w2 + w2O, proc, mp_b2 + bO, proc,
                (int)ROWS_P, D_MODEL, MLPDIM, MLPDIM, D_MODEL, D_MODEL, 0, 0, 0);

    // ===== WRITE: mem += lin_xattn(ln(mem), proc) =====
    ln_kernel<<<(unsigned)ROWS_P, 128, 0, stream>>>(memb, lnb, ln_w_g + bO, ln_w_b + bO, D_MODEL);
    launch_gemm(stream, lnb,  w_q_w + wO, qb, nullptr, nullptr,
                (int)ROWS_P, D_MODEL, D_MODEL, D_MODEL, D_MODEL, D_MODEL, 0, 0, 2);
    launch_gemm(stream, proc, w_k_w + wO, kb, nullptr, nullptr,
                (int)ROWS_P, D_MODEL, D_MODEL, D_MODEL, D_MODEL, D_MODEL, 0, 0, 2);
    launch_gemm(stream, proc, w_v_w + wO, vb, nullptr, nullptr,
                (int)ROWS_P, D_MODEL, D_MODEL, D_MODEL, D_MODEL, D_MODEL, 0, 0, 0);
    launch_gemm(stream, kb, vb, kvb, nullptr, nullptr,
                HD, HD, LPTOK, D_MODEL, D_MODEL, HD, 1, 0, 0,
                B_SZ * NHEADS, NHEADS, sTok, HD, sTok, HD, sKV, (long long)HD * HD);
    ksum_kernel<<<blocks(B_SZ * NHEADS * HD, 256), 256, 0, stream>>>(kb, ksumb, LPTOK);
    launch_gemm(stream, qb, kvb, ob, nullptr, nullptr,
                MEMTOK, HD, HD, D_MODEL, HD, D_MODEL, 0, 0, 0,
                B_SZ * NHEADS, NHEADS, sTok, HD, sKV, (long long)HD * HD, sTok, HD);
    zscale_kernel<<<blocks((long long)B_SZ * NHEADS * MEMTOK, 256), 256, 0, stream>>>(qb, ksumb, ob, MEMTOK);
    launch_gemm(stream, ob, w_o_w + wO, memb, w_o_b + bO, memb,
                (int)ROWS_P, D_MODEL, D_MODEL, D_MODEL, D_MODEL, D_MODEL, 0, 0, 0);

    // ===== MLP (mem) =====
    ln_kernel<<<(unsigned)ROWS_P, 128, 0, stream>>>(memb, lnb, ln_mm_g + bO, ln_mm_b + bO, D_MODEL);
    launch_gemm(stream, lnb, mm_w1 + w1O, big, mm_b1 + b1O, nullptr,
                (int)ROWS_P, MLPDIM, D_MODEL, D_MODEL, MLPDIM, MLPDIM, 0, 0, 1);
    launch_gemm(stream, big, mm_w2 + w2O, memb, mm_b2 + bO, memb,
                (int)ROWS_P, D_MODEL, MLPDIM, MLPDIM, D_MODEL, D_MODEL, 0, 0, 0);
  }

  // ---- decoder ----
  ln_kernel<<<(unsigned)ROWS_P, 128, 0, stream>>>(proc, lnb, norm_g, norm_b, D_MODEL);
  for (int ki = 0; ki < 2; ++ki)
    for (int kj = 0; kj < 2; ++kj) {
      repack_ct<<<blocks(D_MODEL * D_MODEL, 256), 256, 0, stream>>>(dec_ct_w, wpack, ki, kj);
      launch_gemm(stream, lnb, wpack, cttmp, nullptr, nullptr,
                  (int)ROWS_P, D_MODEL, D_MODEL, D_MODEL, D_MODEL, D_MODEL, 0, 0, 0);
      scatter_ct<<<blocks((long long)ROWS_P * D_MODEL, 256), 256, 0, stream>>>(cttmp, yt, dec_ct_b, ki, kj);
    }
  gn_gelu<<<B_SZ * 32, 256, 0, stream>>>(yt, gn_g, gn_b);
  // 1x1 conv: pred = yt @ dec_c_w^T + b   (dec_c_w is [O,I,1,1] -> B in [N,K])
  launch_gemm(stream, yt, dec_c_w, pred, dec_c_b, nullptr,
              B_SZ * TGT_TOK, D_MODEL, D_MODEL, D_MODEL, D_MODEL, D_MODEL, 0, /*bNK=*/1, 0);

  // ---- cosine loss ----
  zero1<<<1, 1, 0, stream>>>(loss);
  cosloss<<<B_SZ * TGT_TOK, 128, 0, stream>>>(pred, target, loss);
}
